// RNN_55439437857102
// MI455X (gfx1250) — matrix operand order repaired and checked by reading, not compiled, not run on hardware
//
#include <hip/hip_runtime.h>
#include <math.h>
#include <stdint.h>

// ---------------------------------------------------------------------------
// Seq2seq GRU (VOCAB=32000, HID=512, LAYERS=2, SEQ=64) for gfx1250 / MI455X.
// f16 WMMA (v_wmma_f32_16x16x32_f16) with f32 accumulate everywhere.
//  K1 gather+cvt embeddings, zero grid-barrier counter
//  K2 convert the 6 recurrent weight matrices to f16
//  K3 GEMM: precompute layer-0 input gates for enc/dec (true WMMA GEMM)
//  K4 persistent 16-WG GRU: recurrent weights TDM-DMA'd into LDS
//     (294KB/WGP, 9 tensor_load_to_lds descriptors), WMMA mat-vecs,
//     atomic grid barrier between dependency phases
//  K5 GEMM: logits[64,32000] = dec_top @ out_W^T + b  (on-the-fly f32->f16,
//     each out_W column read exactly once per block)
//  K6 in-place log_softmax
// CDNA5 paths: v_wmma_f32_16x16x32_f16, tensor_load_to_lds + s_wait_tensorcnt,
// global_load_async_to_lds_b128 + s_wait_asynccnt, global_prefetch_b8,
// ds_load_b128 fragments.
// ---------------------------------------------------------------------------

typedef _Float16 v16h  __attribute__((ext_vector_type(16)));
typedef _Float16 v8h   __attribute__((ext_vector_type(8)));
typedef float    v8f   __attribute__((ext_vector_type(8)));
typedef unsigned u32x4 __attribute__((ext_vector_type(4)));
typedef unsigned u32x8 __attribute__((ext_vector_type(8)));

union F16x16 { v16h v; v8h h[2]; };

#define HID   512
#define SEQT  64
#define G3    1536          // 3*HID
#define NWG   16
#define GRU_THREADS 576     // 18 waves: 3 mats x 6 N-tiles of 16

__device__ __forceinline__ float sigmoidf_(float x) {
    return 1.0f / (1.0f + expf(-x));
}

// ---- async memory->LDS copy of 16B, tracked by ASYNCcnt (no VGPR bounce) --
__device__ __forceinline__ void async_b128(const void* gsrc, void* ldst) {
    unsigned           l = (unsigned)(size_t)ldst;           // low 32 = LDS offset
    unsigned long long g = (unsigned long long)(size_t)gsrc;
    asm volatile("global_load_async_to_lds_b128 %0, %1, off"
                 :: "v"(l), "v"(g) : "memory");
}
__device__ __forceinline__ void async_wait0() {
    asm volatile("s_wait_asynccnt 0x0" ::: "memory");
}

// ---- TDM: DMA a 2-D tile (rows x 512 f16, contiguous rows) into LDS -------
// D# group0: {count=1 | lds_addr | global_addr[56:0] | type=2}
// D# group1: {wg_mask=0, data_size=1(2B), dim0=512, dim1=rows,
//             tile0=512, tile1=rows, dim0_stride=512}
__device__ __forceinline__ void tdm_load_2d(const void* gsrc, void* ldst,
                                            unsigned rows) {
    unsigned long long g = (unsigned long long)(size_t)gsrc;
    u32x4 g0;
    g0[0] = 1u;                                               // count=1 (valid)
    g0[1] = (unsigned)(size_t)ldst;                           // lds_addr (bytes)
    g0[2] = (unsigned)g;                                      // global_addr[31:0]
    g0[3] = (unsigned)((g >> 32) & 0x01FFFFFFu) | 0x80000000u;// addr[56:32]|type=2
    u32x8 g1;
    g1[0] = 1u << 16;                                         // data_size = 2 bytes
    g1[1] = 512u << 16;                                       // tensor_dim0 = 512
    g1[2] = rows << 16;                                       // tensor_dim1 = rows
    g1[3] = 512u << 16;                                       // tile_dim0 = 512
    g1[4] = rows;                                             // tile_dim1 = rows
    g1[5] = 512u;                                             // dim0_stride = 512
    g1[6] = 0u;
    g1[7] = 0u;
    asm volatile("tensor_load_to_lds %0, %1" :: "s"(g0), "s"(g1) : "memory");
}

// ------------------------- grid barrier (monotonic counter) ----------------
__device__ __forceinline__ void grid_barrier(unsigned* cnt, unsigned target) {
    __threadfence();
    __syncthreads();
    if (threadIdx.x == 0) {
        atomicAdd(cnt, 1u);
        while (atomicAdd(cnt, 0u) < target) {
            __builtin_amdgcn_s_sleep(1);
        }
    }
    __syncthreads();
    __threadfence();
}

// ------------------- K1: gather embeddings, zero counter -------------------
__global__ void k_gather(const int* __restrict__ itok, const int* __restrict__ ttok,
                         const float* __restrict__ eemb, const float* __restrict__ demb,
                         _Float16* __restrict__ encx, _Float16* __restrict__ decx,
                         unsigned* __restrict__ cnt) {
    if (blockIdx.x == 0 && threadIdx.x == 0) *cnt = 0u;
    int t  = blockIdx.x;
    int et = itok[t];
    int dt = (t == 0) ? 1 : ttok[t - 1];      // [BOS, target[:-1]]
    for (int k = threadIdx.x; k < HID; k += blockDim.x) {
        encx[t * HID + k] = (_Float16)eemb[(size_t)et * HID + k];
        decx[t * HID + k] = (_Float16)demb[(size_t)dt * HID + k];
    }
}

// ------------------- K2: convert recurrent weights to f16 ------------------
__global__ void k_wcvt(const float* s0, const float* s1, const float* s2,
                       const float* s3, const float* s4, const float* s5,
                       _Float16* __restrict__ dst) {
    const size_t per = (size_t)G3 * HID;
    const float* srcs[6] = { s0, s1, s2, s3, s4, s5 };
    size_t base = ((size_t)blockIdx.x * blockDim.x + threadIdx.x) * 4;
    #pragma unroll
    for (int u = 0; u < 4; ++u) {
        size_t idx = base + u;
        if (idx < 6 * per) {
            int m = (int)(idx / per);
            dst[idx] = (_Float16)srcs[m][idx % per];
        }
    }
}

// ---- K3/K5: C[64,N] = A16[64,512] @ W[N,512]^T + bias ---------------------
// 128-col blocks: 8 waves, wave w owns N-subtile w with 4 M-accumulators,
// so each W column is loaded+converted exactly once per block.
__global__ void k_gemm(const _Float16* __restrict__ A16, const float* __restrict__ W,
                       const float* __restrict__ bias, float* __restrict__ C, int Ntot) {
    extern __shared__ _Float16 ldsA[];          // 64*512 halfs = 64 KB
    for (int i = threadIdx.x; i < (SEQT * HID) / 8; i += blockDim.x)
        async_b128(&A16[i * 8], &ldsA[i * 8]);  // ASYNCcnt path, no VGPR bounce
    async_wait0();
    __syncthreads();

    int wave = threadIdx.x >> 5, lane = threadIdx.x & 31;
    int colg  = blockIdx.x * 128 + wave * 16 + (lane & 15);
    int kofsA = (lane < 16) ? 0 : 8;
    int kofsB = (lane < 16) ? 0 : 16;
    int ar    = lane & 15;

    v8f acc0 = {}, acc1 = {}, acc2 = {}, acc3 = {};
    const float* wr = W + (size_t)colg * HID;
    __builtin_prefetch(wr, 0, 1);               // global_prefetch_b8
    __builtin_prefetch(wr + 256, 0, 1);

    #pragma unroll 4
    for (int kk = 0; kk < 16; ++kk) {
        int k0 = kk * 32;
        F16x16 b;
        #pragma unroll
        for (int i = 0; i < 16; ++i) b.v[i] = (_Float16)wr[k0 + kofsA + i + (i >= 8 ? 8 : 0)];
        F16x16 a0, a1, a2, a3;
        const _Float16* ap = &ldsA[ar * HID + k0 + kofsA];
        a0.h[0] = *(const v8h*)ap;                 a0.h[1] = *(const v8h*)(ap + 16);
        a1.h[0] = *(const v8h*)(ap + 16 * HID);    a1.h[1] = *(const v8h*)(ap + 16 * HID + 16);
        a2.h[0] = *(const v8h*)(ap + 32 * HID);    a2.h[1] = *(const v8h*)(ap + 32 * HID + 16);
        a3.h[0] = *(const v8h*)(ap + 48 * HID);    a3.h[1] = *(const v8h*)(ap + 48 * HID + 16);
        acc0 = __builtin_amdgcn_wmma_f32_16x16x32_f16(false, a0.v, false, b.v, (short)0, acc0, false, false);
        acc1 = __builtin_amdgcn_wmma_f32_16x16x32_f16(false, a1.v, false, b.v, (short)0, acc1, false, false);
        acc2 = __builtin_amdgcn_wmma_f32_16x16x32_f16(false, a2.v, false, b.v, (short)0, acc2, false, false);
        acc3 = __builtin_amdgcn_wmma_f32_16x16x32_f16(false, a3.v, false, b.v, (short)0, acc3, false, false);
    }
    float bv = bias[colg];
    #pragma unroll
    for (int v = 0; v < 8; ++v) {
        int r = (lane < 16) ? v : 8 + v;
        C[(size_t)(r)      * Ntot + colg] = acc0[v] + bv;
        C[(size_t)(16 + r) * Ntot + colg] = acc1[v] + bv;
        C[(size_t)(32 + r) * Ntot + colg] = acc2[v] + bv;
        C[(size_t)(48 + r) * Ntot + colg] = acc3[v] + bv;
    }
}

// ------- WMMA mat-vec: 16 outputs = (16 LDS weight rows) . h  --------------
__device__ __forceinline__ float wmma_gemv16(const _Float16* __restrict__ wrows,
                                             const _Float16* __restrict__ hv, int lane) {
    int n     = lane & 15;
    int kofsA = (lane < 16) ? 0 : 8;
    int kofsB = (lane < 16) ? 0 : 16;
    v8f acc = {};
    #pragma unroll 4
    for (int kk = 0; kk < 16; ++kk) {
        int k0 = kk * 32;
        F16x16 a = {};                           // rows 1..15 zero (unused)
        if (lane == 0 || lane == 16) {
            a.h[0] = *(const v8h*)&hv[k0 + kofsA];
            a.h[1] = *(const v8h*)&hv[k0 + 16 + kofsA];
        }
        F16x16 b;
        const _Float16* wr = &wrows[(size_t)n * HID + k0 + kofsA];
        b.h[0] = *(const v8h*)wr;
        b.h[1] = *(const v8h*)(wr + 16);
        acc = __builtin_amdgcn_wmma_f32_16x16x32_f16(false, a.v, false, b.v,
                                                     (short)0, acc, false, false);
    }
    return acc[0];                               // D row 0, N = lane (lanes 0..15)
}

// ------------------- K4: persistent sequential GRU -------------------------
__global__ void __launch_bounds__(GRU_THREADS)
k_gru(const _Float16* __restrict__ W16,          // [6][1536][512] f16
      const float* __restrict__ gi0_enc, const float* __restrict__ gi0_dec,
      const float* __restrict__ enc_bih, const float* __restrict__ enc_bhh,
      const float* __restrict__ dec_bih, const float* __restrict__ dec_bhh,
      float* __restrict__ h0g, float* __restrict__ h1g,
      _Float16* __restrict__ dtop16, unsigned* __restrict__ cnt) {
    extern __shared__ char smem[];
    _Float16* ldsW = (_Float16*)smem;                       // 3*96*512 halfs
    _Float16* h016 = ldsW + 3 * 96 * HID;                   // 512
    _Float16* h116 = h016 + HID;                            // 512
    float*    gout = (float*)(h116 + HID);                  // 3*96 f32

    const int tid  = threadIdx.x;
    const int j0   = blockIdx.x * 32;                       // owned h slice
    const int wave = tid >> 5, lane = tid & 31;
    const int mat  = wave / 6, tile = wave % 6;
    unsigned bar = 0;

    for (int ph = 0; ph < 2; ++ph) {                        // 0 = enc, 1 = dec
        const _Float16* wsrc = W16 + (size_t)(ph * 3) * G3 * HID;
        const float* gi0 = ph ? gi0_dec : gi0_enc;
        const float* bih = ph ? dec_bih : enc_bih;
        const float* bhh = ph ? dec_bhh : enc_bhh;

        // TDM: DMA this WG's 3 x (3 x 32-row) weight tiles into LDS.
        // Descriptors are wave-uniform; wave 0 issues, all waves sync after.
        __syncthreads();
        if (wave == 0) {
            #pragma unroll
            for (int m = 0; m < 3; ++m) {
                #pragma unroll
                for (int g = 0; g < 3; ++g) {
                    size_t srow = (size_t)m * G3 + g * HID + j0;
                    tdm_load_2d(&wsrc[srow * HID],
                                &ldsW[(size_t)(m * 96 + g * 32) * HID], 32u);
                }
            }
            __builtin_amdgcn_s_wait_tensorcnt(0);
        }
        if (ph == 0 && tid < 32) { h0g[j0 + tid] = 0.f; h1g[j0 + tid] = 0.f; }
        grid_barrier(cnt, NWG * (++bar));

        for (int t = 0; t < SEQT; ++t) {
            // refresh f16 copies of global state
            for (int k = tid; k < HID; k += GRU_THREADS) {
                h016[k] = (_Float16)h0g[k];
                h116[k] = (_Float16)h1g[k];
            }
            __syncthreads();

            // ---- phase 1: gh0 = Whh0 @ h0_prev,  gh1 = Whh1 @ h1_prev ----
            if (mat != 1) {
                const _Float16* hv = (mat == 0) ? h016 : h116;
                float r = wmma_gemv16(ldsW + (size_t)(mat * 96 + tile * 16) * HID, hv, lane);
                if (lane < 16) gout[mat * 96 + tile * 16 + lane] = r;
            }
            __syncthreads();
            if (tid < 32) {                                  // layer-0 elementwise
                int j = j0 + tid;
                float ir = gi0[t * G3 + j];
                float iz = gi0[t * G3 + HID + j];
                float in_ = gi0[t * G3 + 2 * HID + j];
                float hr = gout[0 * 32 + tid] + bhh[0 * HID + j];
                float hz = gout[1 * 32 + tid] + bhh[1 * HID + j];
                float hn = gout[2 * 32 + tid] + bhh[2 * HID + j];
                float r = sigmoidf_(ir + hr), z = sigmoidf_(iz + hz);
                float n = tanhf(in_ + r * hn);
                h0g[j] = (1.f - z) * n + z * h0g[j];
            }
            grid_barrier(cnt, NWG * (++bar));

            // ---- phase 2: gi1 = Wih1 @ h0_new ----
            for (int k = tid; k < HID; k += GRU_THREADS) h016[k] = (_Float16)h0g[k];
            __syncthreads();
            if (mat == 1) {
                float r = wmma_gemv16(ldsW + (size_t)(96 + tile * 16) * HID, h016, lane);
                if (lane < 16) gout[96 + tile * 16 + lane] = r;
            }
            __syncthreads();
            if (tid < 32) {                                  // layer-1 elementwise
                int j = j0 + tid;
                float ir = gout[96 + 0 * 32 + tid]  + bih[G3 + 0 * HID + j];
                float iz = gout[96 + 1 * 32 + tid]  + bih[G3 + 1 * HID + j];
                float in_ = gout[96 + 2 * 32 + tid] + bih[G3 + 2 * HID + j];
                float hr = gout[192 + 0 * 32 + tid] + bhh[G3 + 0 * HID + j];
                float hz = gout[192 + 1 * 32 + tid] + bhh[G3 + 1 * HID + j];
                float hn = gout[192 + 2 * 32 + tid] + bhh[G3 + 2 * HID + j];
                float r = sigmoidf_(ir + hr), z = sigmoidf_(iz + hz);
                float n = tanhf(in_ + r * hn);
                float hnew = (1.f - z) * n + z * h1g[j];
                h1g[j] = hnew;
                if (ph == 1) dtop16[t * HID + j] = (_Float16)hnew;
            }
            grid_barrier(cnt, NWG * (++bar));
        }
    }
}

// ------------------- K6: in-place log_softmax over 32000 -------------------
__global__ void k_lsm(float* __restrict__ C, int Ntot) {
    __shared__ float red[256];
    float* row = C + (size_t)blockIdx.x * Ntot;
    float mx = -3.4e38f;
    for (int i = threadIdx.x; i < Ntot; i += 256) mx = fmaxf(mx, row[i]);
    red[threadIdx.x] = mx; __syncthreads();
    for (int s = 128; s; s >>= 1) {
        if (threadIdx.x < s) red[threadIdx.x] = fmaxf(red[threadIdx.x], red[threadIdx.x + s]);
        __syncthreads();
    }
    mx = red[0]; __syncthreads();
    float sum = 0.f;
    for (int i = threadIdx.x; i < Ntot; i += 256) sum += expf(row[i] - mx);
    red[threadIdx.x] = sum; __syncthreads();
    for (int s = 128; s; s >>= 1) {
        if (threadIdx.x < s) red[threadIdx.x] += red[threadIdx.x + s];
        __syncthreads();
    }
    float lse = mx + logf(red[0]);
    __syncthreads();
    for (int i = threadIdx.x; i < Ntot; i += 256) row[i] -= lse;
}

// ---------------------------------------------------------------------------
extern "C" void kernel_launch(void* const* d_in, const int* in_sizes, int n_in,
                              void* d_out, int out_size, void* d_ws, size_t ws_size,
                              hipStream_t stream) {
    (void)in_sizes; (void)n_in; (void)out_size; (void)ws_size;
    const int*   itok = (const int*)d_in[0];
    const int*   ttok = (const int*)d_in[1];
    const float* eemb = (const float*)d_in[3];
    const float* demb = (const float*)d_in[4];
    const float* eWih = (const float*)d_in[5];
    const float* eWhh = (const float*)d_in[6];
    const float* ebih = (const float*)d_in[7];
    const float* ebhh = (const float*)d_in[8];
    const float* dWih = (const float*)d_in[9];
    const float* dWhh = (const float*)d_in[10];
    const float* dbih = (const float*)d_in[11];
    const float* dbhh = (const float*)d_in[12];
    const float* oW   = (const float*)d_in[13];
    const float* ob   = (const float*)d_in[14];
    float* out = (float*)d_out;

    // workspace carve-up
    char* ws = (char*)d_ws;
    unsigned*  cnt  = (unsigned*)ws;                         // barrier counter
    _Float16*  encx = (_Float16*)(ws + 256);                 // [64][512] f16
    _Float16*  decx = encx + SEQT * HID;
    _Float16*  w16  = decx + SEQT * HID;                     // [6][1536][512] f16
    float*     gi0e = (float*)(w16 + (size_t)6 * G3 * HID);  // [64][1536]
    float*     gi0d = gi0e + SEQT * G3;
    float*     h0   = gi0d + SEQT * G3;                      // [512]
    float*     h1   = h0 + HID;
    _Float16*  dtop = (_Float16*)(h1 + HID);                 // [64][512] f16

    const size_t per = (size_t)G3 * HID;
    const size_t gemmLds = (size_t)SEQT * HID * sizeof(_Float16);           // 64 KB
    const size_t gruLds  = (size_t)(3 * 96 * HID + 2 * HID) * sizeof(_Float16)
                         + 3 * 96 * sizeof(float);                           // ~296 KB

    k_gather<<<SEQT, 256, 0, stream>>>(itok, ttok, eemb, demb, encx, decx, cnt);
    k_wcvt<<<4608, 256, 0, stream>>>(eWhh, eWih + per, eWhh + per,
                                     dWhh, dWih + per, dWhh + per, w16);
    k_gemm<<<G3 / 128, 256, gemmLds, stream>>>(encx, eWih, ebih, gi0e, G3);
    k_gemm<<<G3 / 128, 256, gemmLds, stream>>>(decx, dWih, dbih, gi0d, G3);
    k_gru<<<NWG, GRU_THREADS, gruLds, stream>>>(w16, gi0e, gi0d,
                                                ebih, ebhh, dbih, dbhh,
                                                h0, h1, dtop, cnt);
    k_gemm<<<32000 / 128, 256, gemmLds, stream>>>(dtop, oW, ob, out, 32000);
    k_lsm<<<SEQT, 256, 0, stream>>>(out, 32000);
}
